// Attention_89601607729414
// MI455X (gfx1250) — matrix operand order repaired and checked by reading, not compiled, not run on hardware
//
#include <hip/hip_runtime.h>
#include <hip/hip_bf16.h>

typedef __bf16 bf16;
typedef __attribute__((ext_vector_type(16))) __bf16 v16bf;
typedef __attribute__((ext_vector_type(8)))  float  v8f;
typedef unsigned int v4u __attribute__((ext_vector_type(4)));
typedef int          v8i __attribute__((ext_vector_type(8)));
typedef int          v4i __attribute__((ext_vector_type(4)));

#define B_   4
#define N_   2048
#define D_   768
#define H_   12
#define DH_  64
#define D3_  2304

#if defined(__has_builtin)
#if __has_builtin(__builtin_amdgcn_tensor_load_to_lds) && \
    __has_builtin(__builtin_amdgcn_s_wait_tensorcnt)
#define USE_TDM 1
#endif
#endif
#ifndef USE_TDM
#define USE_TDM 0
#endif

__device__ inline bf16 cvt_bf16(float x) { return (bf16)x; }
__device__ inline bf16 cvt_bf16(bf16 x)  { return x; }
__device__ inline void store_out(float* p, float v) { *p = v; }
__device__ inline void store_out(bf16*  p, float v) { *p = (bf16)v; }

__device__ inline v8f zero8() {
  v8f z = {0.f,0.f,0.f,0.f,0.f,0.f,0.f,0.f};
  return z;
}

union FragCvt { unsigned int u[8]; v16bf v; };

// A-fragment (16x32 bf16, MxK): lane holds row M=lane%16.
// Layout: VGPR v -> K = 16h + 2v + {0,1}
__device__ inline v16bf load_frag_a(const bf16* rowbase, int stride_elems, int lane, int kbase) {
  const unsigned int* r =
      (const unsigned int*)(rowbase + (size_t)(lane & 15) * stride_elems + kbase);
  const int h = (lane >> 4) & 1;
  FragCvt c;
#pragma unroll
  for (int i = 0; i < 8; ++i) c.u[i] = r[h * 8 + i];
  return c.v;
}

// B-fragment (32x16 bf16, KxN) stored in LDS as [N][K] row-major:
// lane holds column N=lane%16; VGPR v -> K = 16h + 2v + {0,1}
__device__ inline v16bf load_frag_b(const bf16* rowbase, int stride_elems, int lane, int kbase) {
  const unsigned int* r =
      (const unsigned int*)(rowbase + (size_t)(lane & 15) * stride_elems + kbase);
  const int h = (lane >> 4) & 1;
  FragCvt c;
#pragma unroll
  for (int i = 0; i < 8; ++i) c.u[i] = r[h * 8 + i];
  return c.v;
}

__device__ inline v8f wmma_bf16(v16bf a, v16bf b, v8f c) {
  return __builtin_amdgcn_wmma_f32_16x16x32_bf16(false, a, false, b, (short)0, c, false, false);
}

#if USE_TDM
// Issue a TDM 2D tile load: tile_w x tile_h bf16 elements, global row stride
// `stride_elems`, DMA'd into LDS at byte offset `lds_off` with 4 dwords of
// padding inserted every 32 dwords (row stride 64 bf16 -> 72 bf16 in LDS).
// D# layout per CDNA5 ISA ch.8 (group0 128b, group1 256b; groups 2/3 zero for 2D).
// This toolchain (clang-23 / amdgpu-toolchain) uses the 6-arg builtin form:
// (uint32x4 g0, int32x8 g1, int32x4, int32x4, int32x8, i32 cpol).
__device__ inline void tdm_load_tile_2d(unsigned lds_off, const bf16* gptr,
                                        int tile_w, int tile_h,
                                        int tensor_h, int stride_elems) {
  const unsigned long long ga = (unsigned long long)(const void*)gptr;
  v4u g0;
  g0[0] = 1u;                                        // count=1 (valid), user mode
  g0[1] = lds_off;                                   // lds_addr (bytes)
  g0[2] = (unsigned)(ga & 0xffffffffu);              // global_addr[31:0]
  g0[3] = (unsigned)((ga >> 32) & 0x01ffffffu)       // global_addr[56:32]
        | (2u << 30);                                // type = 2 ("image")
  v8i g1;
  g1[0] = (1 << 16)                                  // data_size = 2 bytes
        | (1 << 20)                                  // pad_enable
        | (4 << 22)                                  // pad_interval: 32 dwords
        | (3 << 25);                                 // pad_amount: 4 dwords
  g1[1] = (tile_w & 0xffff) << 16;                   // tensor_dim0[15:0]
  g1[2] = ((unsigned)tile_w >> 16)                   // tensor_dim0[31:16]
        | ((tensor_h & 0xffff) << 16);               // tensor_dim1[15:0]
  g1[3] = (((unsigned)tensor_h >> 16) & 0xffff)      // tensor_dim1[31:16]
        | ((tile_w & 0xffff) << 16);                 // tile_dim0
  g1[4] = tile_h & 0xffff;                           // tile_dim1 (tile_dim2 = 0)
  g1[5] = stride_elems;                              // tensor_dim0_stride[31:0]
  g1[6] = 0;                                         // stride0[47:32], stride1[15:0]
  g1[7] = 0;                                         // stride1[47:16]
  v4i z4 = {0, 0, 0, 0};
  v8i z8 = {0, 0, 0, 0, 0, 0, 0, 0};
  __builtin_amdgcn_tensor_load_to_lds(g0, g1, z4, z4, z8, 0);
}
#endif

// ---------------------------------------------------------------------------
// Tiled GEMM: C[M,N] = A[M,K] * B[K,N].
// 128x128 block tile, 256 threads = 8 waves (4 m-waves x 2 n-waves);
// each wave computes a 32x64 strip = 8 f32 WMMA accumulators per K-step.
// Register double-buffering + global_prefetch one tile ahead.
// ---------------------------------------------------------------------------
template <typename TA, typename TB, typename TO>
__global__ __launch_bounds__(256)
void wmma_gemm_kernel(const TA* __restrict__ A, const TB* __restrict__ B,
                      TO* __restrict__ C, int M, int N, int K) {
  __shared__ bf16 sA[128][40];  // [m][k], padded (20 dwords/row -> conflict free)
  __shared__ bf16 sB[128][40];  // [n][k]

  const int tid  = threadIdx.x;
  const int lane = tid & 31;
  const int wave = tid >> 5;
  const int wm = wave & 3;   // 4 m-waves, 32 rows each
  const int wn = wave >> 2;  // 2 n-waves, 64 cols each
  const int n0 = blockIdx.x * 128;
  const int m0 = blockIdx.y * 128;

  const int arow = tid >> 1;          // 0..127
  const int acb  = (tid & 1) * 16;    // 0 or 16
  const int bk   = tid & 31;          // 0..31
  const int bnb  = (tid >> 5) * 16;   // 0..112

  TA ra[16];
  TB rb[16];
  auto load_regs = [&](int kk) {
    const TA* ga = A + (size_t)(m0 + arow) * K + kk + acb;
#pragma unroll
    for (int i = 0; i < 16; ++i) ra[i] = ga[i];
    const TB* gb = B + (size_t)(kk + bk) * N + n0 + bnb;
#pragma unroll
    for (int i = 0; i < 16; ++i) rb[i] = gb[i];
  };
  auto store_lds = [&]() {
#pragma unroll
    for (int i = 0; i < 16; ++i) sA[arow][acb + i] = cvt_bf16(ra[i]);
#pragma unroll
    for (int i = 0; i < 16; ++i) sB[bnb + i][bk] = cvt_bf16(rb[i]);
  };

  v8f acc[2][4];
#pragma unroll
  for (int mt = 0; mt < 2; ++mt)
#pragma unroll
    for (int nt = 0; nt < 4; ++nt) acc[mt][nt] = zero8();

  load_regs(0);
  store_lds();
  __syncthreads();

  for (int kk = 32; kk <= K; kk += 32) {
    if (kk < K) {
      load_regs(kk);  // overlap global latency with WMMA below
      if (kk + 32 < K) {
        __builtin_prefetch(A + (size_t)(m0 + arow) * K + kk + 32 + acb, 0, 1);
        __builtin_prefetch(B + (size_t)(kk + 32 + bk) * N + n0 + bnb, 0, 1);
      }
    }

    v16bf af0 = load_frag_a(&sA[wm * 32][0],      40, lane, 0);
    v16bf af1 = load_frag_a(&sA[wm * 32 + 16][0], 40, lane, 0);
#pragma unroll
    for (int nt = 0; nt < 4; ++nt) {
      v16bf bfr = load_frag_b(&sB[wn * 64 + nt * 16][0], 40, lane, 0);
      acc[0][nt] = wmma_bf16(af0, bfr, acc[0][nt]);
      acc[1][nt] = wmma_bf16(af1, bfr, acc[1][nt]);
    }
    __syncthreads();
    if (kk < K) {
      store_lds();
      __syncthreads();
    }
  }

  const int h = (lane >> 4) & 1;
#pragma unroll
  for (int mt = 0; mt < 2; ++mt)
#pragma unroll
    for (int nt = 0; nt < 4; ++nt)
#pragma unroll
      for (int r = 0; r < 8; ++r) {
        const int row = m0 + wm * 32 + mt * 16 + r + 8 * h;  // C: M = r + 8*half
        const int col = n0 + wn * 64 + nt * 16 + (lane & 15);
        store_out(&C[(size_t)row * N + col], acc[mt][nt][r]);
      }
}

// ---------------------------------------------------------------------------
// Flash attention: one block = (b, h, 64-query tile); 4 waves, 16 q-rows each.
// Q/K tiles staged by the Tensor Data Mover (TENSORcnt), V transposed manually.
// ---------------------------------------------------------------------------
__global__ __launch_bounds__(128)
void flash_attn_kernel(const bf16* __restrict__ qkv, bf16* __restrict__ out) {
  __shared__ bf16 sQ[64][72];       // [q][dh]
  __shared__ bf16 sK[64][72];       // [key][dh]  (== B-layout [N][K] for QK^T)
  __shared__ bf16 sVt[64][72];      // [dh][key]  (== B-layout [N][K] for P*V)
  __shared__ bf16 sP[4][16][72];    // per-wave probs [q][key]

  const int tid  = threadIdx.x;
  const int lane = tid & 31;
  const int wave = tid >> 5;
  const int h16  = (lane >> 4) & 1;
  const int bh   = blockIdx.x;
  const int b    = bh / H_;
  const int hh   = bh % H_;
  const int qtile = blockIdx.y;
  const int q0    = qtile * 64;

  const bf16* qbase = qkv + (size_t)(b * N_) * D3_ + hh * DH_;
  const bf16* kbase = qbase + D_;
  const bf16* vbase = qbase + 2 * D_;

  const int srow = tid >> 1;         // staging row (0..63)
  const int scb  = (tid & 1) * 32;   // staging col base (0 or 32)

#if USE_TDM
  if (wave == 0) {  // TDM: DMA Q tile into LDS (hardware pads rows to 72 bf16)
    tdm_load_tile_2d((unsigned)(size_t)(void*)&sQ[0][0],
                     qbase + (size_t)q0 * D3_, DH_, 64, N_, D3_);
    __builtin_amdgcn_s_wait_tensorcnt(0);
  }
#else
  {
    const bf16* g = qbase + (size_t)(q0 + srow) * D3_ + scb;
#pragma unroll
    for (int i = 0; i < 32; ++i) sQ[srow][scb + i] = g[i];
  }
#endif
  __syncthreads();

  float m_run[8], l_run[8];
#pragma unroll
  for (int r = 0; r < 8; ++r) { m_run[r] = -__builtin_inff(); l_run[r] = 0.f; }
  v8f o[4];
#pragma unroll
  for (int i = 0; i < 4; ++i) o[i] = zero8();

  const v16bf qf0 = load_frag_a(&sQ[wave * 16][0], 72, lane, 0);
  const v16bf qf1 = load_frag_a(&sQ[wave * 16][0], 72, lane, 32);

  for (int kt = 0; kt <= qtile; ++kt) {
#if USE_TDM
    if (wave == 0) {  // TDM: DMA K tile into LDS
      tdm_load_tile_2d((unsigned)(size_t)(void*)&sK[0][0],
                       kbase + (size_t)(kt * 64) * D3_, DH_, 64, N_, D3_);
    }
#else
    {
      const bf16* gk = kbase + (size_t)(kt * 64 + srow) * D3_ + scb;
#pragma unroll
      for (int i = 0; i < 32; ++i) sK[srow][scb + i] = gk[i];
    }
#endif
    {  // stage transposed V tile (TDM cannot transpose)
      const bf16* gv = vbase + (size_t)(kt * 64 + srow) * D3_ + scb;
#pragma unroll
      for (int i = 0; i < 32; ++i) sVt[scb + i][srow] = gv[i];
    }
    if (kt < qtile) {  // prefetch next K/V tile into cache
      __builtin_prefetch(kbase + (size_t)((kt + 1) * 64 + srow) * D3_ + scb, 0, 1);
      __builtin_prefetch(vbase + (size_t)((kt + 1) * 64 + srow) * D3_ + scb, 0, 1);
    }
#if USE_TDM
    if (wave == 0) __builtin_amdgcn_s_wait_tensorcnt(0);
#endif
    __syncthreads();

    // --- S = Q K^T / sqrt(dh), causal mask ---
    float st[4][8];
#pragma unroll
    for (int kc = 0; kc < 4; ++kc) {
      v8f s = zero8();
      v16bf kf0 = load_frag_b(&sK[kc * 16][0], 72, lane, 0);
      v16bf kf1 = load_frag_b(&sK[kc * 16][0], 72, lane, 32);
      s = wmma_bf16(qf0, kf0, s);
      s = wmma_bf16(qf1, kf1, s);
#pragma unroll
      for (int r = 0; r < 8; ++r) {
        const int qg = q0 + wave * 16 + r + 8 * h16;
        const int kg = kt * 64 + kc * 16 + (lane & 15);
        const float v = s[r] * 0.125f;  // 1/sqrt(64)
        st[kc][r] = (kg > qg) ? -__builtin_inff() : v;
      }
    }

    // --- online softmax (row reductions across each 16-lane half) ---
    float alpha[8];
#pragma unroll
    for (int r = 0; r < 8; ++r) {
      float rowmax = st[0][r];
#pragma unroll
      for (int kc = 1; kc < 4; ++kc) rowmax = fmaxf(rowmax, st[kc][r]);
#pragma unroll
      for (int off = 1; off <= 8; off <<= 1)
        rowmax = fmaxf(rowmax, __shfl_xor(rowmax, off, 32));
      const float mnew = fmaxf(m_run[r], rowmax);
      alpha[r] = __expf(m_run[r] - mnew);
      float rs = 0.f;
#pragma unroll
      for (int kc = 0; kc < 4; ++kc) {
        const float p = __expf(st[kc][r] - mnew);
        st[kc][r] = p;
        rs += p;
      }
#pragma unroll
      for (int off = 1; off <= 8; off <<= 1)
        rs += __shfl_xor(rs, off, 32);
      l_run[r] = l_run[r] * alpha[r] + rs;
      m_run[r] = mnew;
    }
#pragma unroll
    for (int nt = 0; nt < 4; ++nt)
#pragma unroll
      for (int r = 0; r < 8; ++r) o[nt][r] = o[nt][r] * alpha[r];

    // --- write P (C-layout -> LDS), re-read as A-fragments ---
#pragma unroll
    for (int kc = 0; kc < 4; ++kc)
#pragma unroll
      for (int r = 0; r < 8; ++r)
        sP[wave][r + 8 * h16][kc * 16 + (lane & 15)] = cvt_bf16(st[kc][r]);

    // --- O += P V ---
#pragma unroll
    for (int ks = 0; ks < 2; ++ks) {
      v16bf pf = load_frag_a(&sP[wave][0][0], 72, lane, ks * 32);
#pragma unroll
      for (int nt = 0; nt < 4; ++nt) {
        v16bf vf = load_frag_b(&sVt[nt * 16][0], 72, lane, ks * 32);
        o[nt] = wmma_bf16(pf, vf, o[nt]);
      }
    }
    __syncthreads();
  }

  // epilogue: normalize and write merged-head layout [b, n, d] (bf16)
#pragma unroll
  for (int nt = 0; nt < 4; ++nt)
#pragma unroll
    for (int r = 0; r < 8; ++r) {
      const int qg  = q0 + wave * 16 + r + 8 * h16;
      const int col = hh * DH_ + nt * 16 + (lane & 15);
      const float val = o[nt][r] / l_run[r];
      out[(size_t)(b * N_ + qg) * D_ + col] = cvt_bf16(val);
    }
}

// ---------------------------------------------------------------------------
extern "C" void kernel_launch(void* const* d_in, const int* in_sizes, int n_in,
                              void* d_out, int out_size, void* d_ws, size_t ws_size,
                              hipStream_t stream) {
  (void)in_sizes; (void)n_in; (void)out_size; (void)ws_size;

  const float* x     = (const float*)d_in[0];
  // d_in[1] = causal mask (hardcoded in-kernel)
  const float* w_qkv = (const float*)d_in[2];
  const float* w_out = (const float*)d_in[3];
  float* out = (float*)d_out;

  bf16* qkv  = (bf16*)d_ws;                                        // [8192][2304] bf16
  bf16* attn = (bf16*)((char*)d_ws + (size_t)B_ * N_ * D3_ * 2);   // [8192][768]  bf16

  // 1) QKV projection: x(f32) @ w_qkv(f32) -> qkv(bf16)
  wmma_gemm_kernel<float, float, bf16>
      <<<dim3(D3_ / 128, (B_ * N_) / 128), 256, 0, stream>>>(x, w_qkv, qkv, B_ * N_, D3_, D_);
  // 2) causal flash attention -> attn(bf16) in merged [b,n,d] layout
  flash_attn_kernel<<<dim3(B_ * H_, N_ / 64), 128, 0, stream>>>(qkv, attn);
  // 3) output projection: attn(bf16) @ w_out(f32) -> out(f32)
  wmma_gemm_kernel<bf16, float, float>
      <<<dim3(D_ / 128, (B_ * N_) / 128), 256, 0, stream>>>(attn, w_out, out, B_ * N_, D_, D_);
}
